// AttentionLM_89060441850513
// MI455X (gfx1250) — matrix operand order repaired and checked by reading, not compiled, not run on hardware
//
#include <hip/hip_runtime.h>
#include <hip/hip_bf16.h>

typedef _Float16 half8  __attribute__((ext_vector_type(8)));
typedef _Float16 half16 __attribute__((ext_vector_type(16)));
typedef float    f32x8  __attribute__((ext_vector_type(8)));

#define BB    2
#define SS    2048
#define VOC   32000
#define EMB   1024
#define NH    16
#define HD    64
#define BS    (BB*SS)

// ---------------- WMMA helpers (gfx1250 wave32, 16x16x32 f16 -> f32) ----------------

static __device__ __forceinline__ f32x8 wmma16(half16 a, half16 b, f32x8 c) {
  // 8 args: (neg_a, A, neg_b, B, c_mod, C, reuse_a, reuse_b)
  return __builtin_amdgcn_wmma_f32_16x16x32_f16(false, a, false, b, (short)0, c, false, false);
}

// A-matrix 16x32 f16 fragment: lane l holds row M=l&15;
// halves 0..7 = K=klo+0..7, halves 8..15 = K=klo+8..15, klo=(l>>4)*16.
// Two contiguous 16B chunks per lane.
static __device__ __forceinline__ half16 load_a_frag(const _Float16* tile, int row_stride, int lane) {
  int m   = lane & 15;
  int klo = (lane >> 4) * 16;
  half8 lo = *(const half8*)(tile + (size_t)m * row_stride + klo);
  half8 hi = *(const half8*)(tile + (size_t)m * row_stride + klo + 8);
  return __builtin_shufflevector(lo, hi, 0,1,2,3,4,5,6,7,8,9,10,11,12,13,14,15);
}

// B-matrix 32x16 f16 fragment from K-MAJOR storage (stride between N-columns = n_stride):
// lane l holds N=l&15; half j = K=16*(l>>4)+j -> one contiguous 32B load per lane.
static __device__ __forceinline__ half16 load_b_frag(const _Float16* base, int n_stride, int lane) {
  int n  = lane & 15;
  int kb = (lane >> 4) * 16;
  return *(const half16*)(base + (size_t)n * n_stride + kb);
}

// ---------------- K0a: fp32 [H,E,D] -> f16 [H,D,E] (K-major heads) ----------------

__global__ void cvt_qkv_w(const float* __restrict__ src, _Float16* __restrict__ dst) {
  // one block per (h, d): thread walks e (write-contiguous)
  int h = blockIdx.y;
  int d = blockIdx.x;
  const float* s = src + (size_t)h * EMB * HD + d;
  _Float16*    o = dst + ((size_t)h * HD + d) * EMB;
  for (int e = threadIdx.x; e < EMB; e += blockDim.x)
    o[e] = (_Float16)s[(size_t)e * HD];
}

// ---------------- K0b: fp32 lin_w [E,V] -> f16 [V,E] ----------------

__global__ void cvt_lin_w(const float* __restrict__ src, _Float16* __restrict__ dst) {
  // one block per vocab column n: thread walks k (write-contiguous)
  int n = blockIdx.x;
  _Float16* o = dst + (size_t)n * EMB;
  for (int k = threadIdx.x; k < EMB; k += blockDim.x)
    o[k] = (_Float16)src[(size_t)k * VOC + n];
}

// ---------------- K1: embedding + positional + exact GeLU -> f16 ----------------

__global__ void embed_gelu(const int* __restrict__ x, const float* __restrict__ et,
                           const float* __restrict__ pt, _Float16* __restrict__ h) {
  int row = blockIdx.x;               // 0..BS-1
  int s   = row & (SS - 1);
  int idx = x[row];
  const float* erow = et + (size_t)idx * EMB;
  const float* prow = pt + (size_t)s * EMB;
  for (int t = threadIdx.x; t < EMB; t += blockDim.x) {
    float v = erow[t] + prow[t];
    float g = 0.5f * v * (1.0f + erff(v * 0.70710678118654752f));
    h[(size_t)row * EMB + t] = (_Float16)g;
  }
}

// ---------------- K2: QKV projections, one wave -> 16x64 tile ----------------
// grid = (BS/16, NH, 3), block = 32.  Weights are K-major [H,D,E].
// Q,K stored [B,H,S,D]; V stored transposed [B,H,D,S] for K-major P@V fragments.

__global__ void qkv_proj(const _Float16* __restrict__ h,
                         const _Float16* __restrict__ wqT, const _Float16* __restrict__ wkT,
                         const _Float16* __restrict__ wvT,
                         _Float16* __restrict__ q, _Float16* __restrict__ k, _Float16* __restrict__ v) {
  int lane  = threadIdx.x;
  int m0    = blockIdx.x * 16;
  int head  = blockIdx.y;
  int which = blockIdx.z;
  const _Float16* w = (which == 0 ? wqT : which == 1 ? wkT : wvT) + (size_t)head * HD * EMB;
  _Float16*       o = (which == 0 ? q   : which == 1 ? k   : v);

  f32x8 acc[4] = {{}, {}, {}, {}};
  for (int k0 = 0; k0 < EMB; k0 += 32) {
    half16 a = load_a_frag(h + (size_t)m0 * EMB + k0, EMB, lane);
    #pragma unroll
    for (int nt = 0; nt < 4; ++nt) {
      half16 b = load_b_frag(w + (size_t)(nt * 16) * EMB + k0, EMB, lane);
      acc[nt] = wmma16(a, b, acc[nt]);
    }
  }
  #pragma unroll
  for (int nt = 0; nt < 4; ++nt) {
    int n = nt * 16 + (lane & 15);
    #pragma unroll
    for (int r = 0; r < 8; ++r) {
      int m   = r + 8 * (lane >> 4);
      int row = m0 + m;
      int b   = row >> 11;            // /SS
      int s   = row & (SS - 1);
      if (which == 2)                 // V transposed: [B,H,D,S]
        o[(((size_t)(b * NH + head)) * HD + n) * SS + s] = (_Float16)acc[nt][r];
      else                            // Q,K: [B,H,S,D]
        o[(((size_t)(b * NH + head)) * SS + s) * HD + n] = (_Float16)acc[nt][r];
    }
  }
}

// ---------------- K3: flash attention per (b,h), one wave -> 16 query rows ----------------
// grid = (SS/16, NH, BB), block = 32.  scores = qk/64 (faithful double scaling),
// softmax (no mask), z = attn @ v.  V is [B,H,D,S] so P@V B-fragments are contiguous.

__global__ void flash_attn(const _Float16* __restrict__ q, const _Float16* __restrict__ kk,
                           const _Float16* __restrict__ vT, _Float16* __restrict__ z) {
  __shared__ _Float16 ptile[16 * 32];

  int lane = threadIdx.x;
  int s0   = blockIdx.x * 16;
  int hd   = blockIdx.y;
  int b    = blockIdx.z;
  size_t base = ((size_t)(b * NH + hd)) * SS * HD;
  const _Float16* qb = q  + base;
  const _Float16* kb = kk + base;
  const _Float16* vb = vT + base;     // [D, S] within this head

  // Q fragments: 16x64 split into two 16x32 A fragments (K = D)
  half16 aq[2];
  #pragma unroll
  for (int kc = 0; kc < 2; ++kc)
    aq[kc] = load_a_frag(qb + (size_t)s0 * HD + kc * 32, HD, lane);

  f32x8 acc[4] = {{}, {}, {}, {}};
  float row_max[8], row_sum[8], alpha[8];
  #pragma unroll
  for (int r = 0; r < 8; ++r) { row_max[r] = -INFINITY; row_sum[r] = 0.0f; }

  const float scale = 1.0f / 64.0f;

  for (int kt = 0; kt < SS; kt += 32) {
    // two 16x16 score tiles; B = K^T is D-contiguous in [S,D] storage
    f32x8 sc[2];
    #pragma unroll
    for (int tt = 0; tt < 2; ++tt) {
      f32x8 c = {};
      #pragma unroll
      for (int kc = 0; kc < 2; ++kc) {
        half16 bkf = load_b_frag(kb + (size_t)(kt + tt * 16) * HD + kc * 32, HD, lane);
        c = wmma16(aq[kc], bkf, c);
      }
      sc[tt] = c;
    }

    // online softmax; row m = r + 8*(lane>>4) lives in a 16-lane half
    #pragma unroll
    for (int r = 0; r < 8; ++r) {
      float v0 = sc[0][r] * scale, v1 = sc[1][r] * scale;
      float lm = fmaxf(v0, v1);
      #pragma unroll
      for (int msk = 1; msk < 16; msk <<= 1) lm = fmaxf(lm, __shfl_xor(lm, msk, 32));
      float nm = fmaxf(row_max[r], lm);
      alpha[r] = expf(row_max[r] - nm);
      float p0 = expf(v0 - nm), p1 = expf(v1 - nm);
      float ps = p0 + p1;
      #pragma unroll
      for (int msk = 1; msk < 16; msk <<= 1) ps += __shfl_xor(ps, msk, 32);
      row_sum[r] = row_sum[r] * alpha[r] + ps;
      row_max[r] = nm;
      sc[0][r] = p0; sc[1][r] = p1;
    }
    #pragma unroll
    for (int nt = 0; nt < 4; ++nt)
      #pragma unroll
      for (int r = 0; r < 8; ++r) acc[nt][r] *= alpha[r];

    // redistribute P from C-layout to A-layout via LDS
    #pragma unroll
    for (int r = 0; r < 8; ++r) {
      int m = r + 8 * (lane >> 4);
      ptile[m * 32 + (lane & 15)]      = (_Float16)sc[0][r];
      ptile[m * 32 + 16 + (lane & 15)] = (_Float16)sc[1][r];
    }
    __syncthreads();
    half16 pf = load_a_frag(ptile, 32, lane);
    __syncthreads();

    // acc += P (16x32) x V (32x64); V^T is S-contiguous -> contiguous B fragments
    #pragma unroll
    for (int nt = 0; nt < 4; ++nt) {
      half16 bv = load_b_frag(vb + (size_t)(nt * 16) * SS + kt, SS, lane);
      acc[nt] = wmma16(pf, bv, acc[nt]);
    }
  }

  // z stored as [B, S, H*D] f16 for the final GEMM
  #pragma unroll
  for (int nt = 0; nt < 4; ++nt) {
    int d = nt * 16 + (lane & 15);
    #pragma unroll
    for (int r = 0; r < 8; ++r) {
      int m = r + 8 * (lane >> 4);
      float val = acc[nt][r] / row_sum[r];
      z[((size_t)(b * SS + s0 + m)) * EMB + hd * HD + d] = (_Float16)val;
    }
  }
}

// ---------------- K4: logits = relu(Z @ W + b), one wave -> 16x128 tile ----------------
// grid = (VOC/128, BS/16), block = 32.  W is K-major [V,E]: every B fragment is one
// contiguous 32B load; A fragment reused 8x per K-step.

__global__ void logits_gemm(const _Float16* __restrict__ z, const _Float16* __restrict__ wT,
                            const float* __restrict__ bias, float* __restrict__ out) {
  int lane = threadIdx.x;
  int n0   = blockIdx.x * 128;
  int m0   = blockIdx.y * 16;

  f32x8 acc[8] = {{}, {}, {}, {}, {}, {}, {}, {}};
  for (int k0 = 0; k0 < EMB; k0 += 32) {
    // prefetch next K-panel of the weight rows this wave will read (global_prefetch_b8)
    if (k0 + 32 < EMB)
      __builtin_prefetch(wT + (size_t)(n0 + lane * 4) * EMB + k0 + 32, 0, 1);
    half16 a = load_a_frag(z + (size_t)m0 * EMB + k0, EMB, lane);
    #pragma unroll
    for (int nt = 0; nt < 8; ++nt) {
      half16 b = load_b_frag(wT + (size_t)(n0 + nt * 16) * EMB + k0, EMB, lane);
      acc[nt] = wmma16(a, b, acc[nt]);
    }
  }
  #pragma unroll
  for (int nt = 0; nt < 8; ++nt) {
    int n = n0 + nt * 16 + (lane & 15);
    float bv = bias[n];
    #pragma unroll
    for (int r = 0; r < 8; ++r) {
      int m = m0 + r + 8 * (lane >> 4);
      out[(size_t)m * VOC + n] = fmaxf(acc[nt][r] + bv, 0.0f);
    }
  }
}

// ---------------- launch ----------------

extern "C" void kernel_launch(void* const* d_in, const int* in_sizes, int n_in,
                              void* d_out, int out_size, void* d_ws, size_t ws_size,
                              hipStream_t stream) {
  (void)in_sizes; (void)n_in; (void)out_size; (void)ws_size;
  const int*   x  = (const int*)d_in[0];
  const float* et = (const float*)d_in[1];
  const float* pt = (const float*)d_in[2];
  const float* wq = (const float*)d_in[3];
  const float* wk = (const float*)d_in[4];
  const float* wv = (const float*)d_in[5];
  const float* lw = (const float*)d_in[6];
  const float* lb = (const float*)d_in[7];
  float* out = (float*)d_out;

  char* ws = (char*)d_ws;
  size_t off = 0;
  auto carve = [&](size_t elems) {
    _Float16* p = (_Float16*)(ws + off);
    off = (off + elems * sizeof(_Float16) + 255) & ~(size_t)255;
    return p;
  };
  _Float16* h16  = carve((size_t)BS * EMB);
  _Float16* wqT  = carve((size_t)NH * HD * EMB);
  _Float16* wkT  = carve((size_t)NH * HD * EMB);
  _Float16* wvT  = carve((size_t)NH * HD * EMB);
  _Float16* q16  = carve((size_t)BB * NH * SS * HD);
  _Float16* k16  = carve((size_t)BB * NH * SS * HD);
  _Float16* v16T = carve((size_t)BB * NH * HD * SS);
  _Float16* z16  = carve((size_t)BS * EMB);
  _Float16* wT   = carve((size_t)VOC * EMB);

  cvt_qkv_w<<<dim3(HD, NH), 256, 0, stream>>>(wq, wqT);
  cvt_qkv_w<<<dim3(HD, NH), 256, 0, stream>>>(wk, wkT);
  cvt_qkv_w<<<dim3(HD, NH), 256, 0, stream>>>(wv, wvT);
  cvt_lin_w<<<VOC, 256, 0, stream>>>(lw, wT);

  embed_gelu<<<BS, 256, 0, stream>>>(x, et, pt, h16);

  qkv_proj<<<dim3(BS / 16, NH, 3), 32, 0, stream>>>(h16, wqT, wkT, wvT, q16, k16, v16T);

  flash_attn<<<dim3(SS / 16, NH, BB), 32, 0, stream>>>(q16, k16, v16T, z16);

  logits_gemm<<<dim3(VOC / 128, BS / 16), 32, 0, stream>>>(z16, wT, lb, out);
}
